// LinearAttention_39092792328968
// MI455X (gfx1250) — matrix operand order repaired and checked by reading, not compiled, not run on hardware
//
#include <hip/hip_runtime.h>

// ---------------------------------------------------------------------------
// Problem constants (from reference)
// ---------------------------------------------------------------------------
static constexpr int Bv   = 2;
static constexpr int Sv   = 4096;
static constexpr int Dv   = 2048;
static constexpr int HQv  = 32;
static constexpr int HKv  = 8;
static constexpr int Hv   = 32;
static constexpr int DKv  = 64;
static constexpr int DVv  = 64;
static constexpr int OUTv = 2048;
static constexpr int CH   = 64;
static constexpr int NCv  = Sv / CH;                       // 64
static constexpr int QKVv = HQv*DKv + HKv*DKv + HKv*DVv;   // 3072

typedef __attribute__((ext_vector_type(16))) __bf16 v16bf;
typedef __attribute__((ext_vector_type(8)))  float  v8f;
typedef __attribute__((ext_vector_type(4)))  int    v4i;

union BFrag { v16bf v; unsigned short u[16]; uint4 q[2]; };
union AccF  { v8f   v; float          f[8];  };

// fast f32 -> bf16 (round-half-up): 1 add + shift
static __device__ __forceinline__ unsigned short f2u(float f) {
    return (unsigned short)((__builtin_bit_cast(unsigned, f) + 0x8000u) >> 16);
}
// pack two f32 -> packed bf16x2 with a single v_perm_b32
static __device__ __forceinline__ unsigned pk2(float lo, float hi) {
    unsigned ulo = __builtin_bit_cast(unsigned, lo) + 0x8000u;
    unsigned uhi = __builtin_bit_cast(unsigned, hi) + 0x8000u;
    return __builtin_amdgcn_perm(uhi, ulo, 0x07060302u);
}
static __device__ __forceinline__ float u2f(unsigned short h) {
    return __builtin_bit_cast(float, ((unsigned)h) << 16);
}

// ---------------------------------------------------------------------------
// 16-byte global -> LDS copy. Uses the CDNA5 async-to-LDS path (ASYNCcnt)
// when the toolchain exposes it; otherwise a plain b128 load/store pair.
// Builtin signature (from hipcc diagnostic): param0 is AS1 int4*, 4 args.
// ---------------------------------------------------------------------------
#if __has_builtin(__builtin_amdgcn_global_load_async_to_lds_b128)
#define ATHENA_ASYNC_LDS 1
typedef __attribute__((address_space(1))) v4i* gas1_t;
typedef __attribute__((address_space(3))) v4i* las3_t;
#else
#define ATHENA_ASYNC_LDS 0
#endif

static __device__ __forceinline__ void cp16(void* lds, const void* g) {
#if ATHENA_ASYNC_LDS
    __builtin_amdgcn_global_load_async_to_lds_b128((gas1_t)g, (las3_t)lds, 0, 0);
#else
    *(uint4*)lds = *(const uint4*)g;
#endif
}
static __device__ __forceinline__ void cp_wait() {
#if ATHENA_ASYNC_LDS
#if __has_builtin(__builtin_amdgcn_s_wait_asynccnt)
    __builtin_amdgcn_s_wait_asynccnt(0);
#else
    asm volatile("s_wait_asynccnt 0x0" ::: "memory");
#endif
#endif
}

// A-layout fragment (lane = row M): K = [8h,8h+8) u [16+8h,16+8h+8)
static __device__ __forceinline__ void loadA(BFrag& f, const unsigned short* row, int half) {
    f.q[0] = *(const uint4*)(row + 8 * half);
    f.q[1] = *(const uint4*)(row + 16 + 8 * half);
}
// B-layout fragment (lane = col N, K-contiguous row): K = [8h,8h+8) u [16+8h,16+8h+8)
static __device__ __forceinline__ void loadB(BFrag& f, const unsigned short* row, int half) {
    f.q[0] = *(const uint4*)(row + 8 * half);
    f.q[1] = *(const uint4*)(row + 16 + 8 * half);
}

static __device__ __forceinline__ v8f wmma_bf16(const BFrag& a, const BFrag& b, v8f c) {
    return __builtin_amdgcn_wmma_f32_16x16x32_bf16(false, a.v, false, b.v,
                                                   (short)0, c, false, false);
}

// ---------------------------------------------------------------------------
// Transpose+convert weights: in f32 [R][C] -> out bf16 [C][R].
// ---------------------------------------------------------------------------
__global__ __launch_bounds__(256)
void transpose_w(const float* __restrict__ in, unsigned short* __restrict__ out,
                 int R, int C)
{
    __shared__ alignas(16) unsigned short t[32 * 33];
    const int tid = threadIdx.x;
    const int c0 = blockIdx.x * 32, r0 = blockIdx.y * 32;
    for (int i = tid; i < 1024; i += 256) {
        int r = i >> 5, c = i & 31;
        t[r * 33 + c] = f2u(in[(size_t)(r0 + r) * C + c0 + c]);
    }
    __syncthreads();
    for (int i = tid; i < 512; i += 256) {
        int c = i >> 4, r2 = (i & 15) << 1;
        unsigned u = (unsigned)t[r2 * 33 + c] | ((unsigned)t[(r2 + 1) * 33 + c] << 16);
        *(unsigned*)(out + (size_t)(c0 + c) * R + r0 + r2) = u;
    }
}

// ---------------------------------------------------------------------------
// Tiled GEMM: C[M,N] = A[M,K] * Bt^T, Bt is bf16 [N][K].
// AMODE 0: A f32 (converted at stage). AMODE 1: A bf16 (async-copied).
// CMODE 0: C bf16. CMODE 1: C f32.
// Block tile 256x128x32, double-buffered LDS, 256 threads = 8 waves (4x2),
// wave tile 64x64 = 4x4 WMMA fragments (16 WMMAs per K-tile per wave).
// ---------------------------------------------------------------------------
template<int AMODE, int CMODE>
__global__ __launch_bounds__(256)
void gemm_wmma(const void* __restrict__ Ap, const unsigned short* __restrict__ Bt,
               void* __restrict__ Cp, int M, int N, int K)
{
    constexpr int BM = 256, BN = 128, BK = 32;
    __shared__ alignas(16) unsigned short As[2][BM * BK];   // [m][k]
    __shared__ alignas(16) unsigned short Bs[2][BN * BK];   // [n][k]

    const int tid  = threadIdx.x;
    const int n0   = blockIdx.x * BN;
    const int m0   = blockIdx.y * BM;
    const int w    = tid >> 5, lane = tid & 31;
    const int half = lane >> 4, lq  = lane & 15;
    const int wr   = w >> 1,   wc   = w & 1;    // 4 wave-rows x 2 wave-cols

    auto stage = [&](int k0, int bf) {
        if (AMODE == 0) {
            const float* A = (const float*)Ap;
            for (int i = tid; i < BM * BK / 4; i += 256) {
                int r = i >> 3, c4 = (i & 7) << 2;
                float4 t = *(const float4*)(A + (size_t)(m0 + r) * K + k0 + c4);
                uint2 p; p.x = pk2(t.x, t.y); p.y = pk2(t.z, t.w);
                *(uint2*)(&As[bf][r * BK + c4]) = p;
            }
            if (k0 + BK < K)   // cover the f32 A stream (no async path for it)
                __builtin_prefetch(A + (size_t)(m0 + (tid >> 1) * 2) * K + k0 + BK, 0, 1);
        } else {
            const unsigned short* A = (const unsigned short*)Ap;
            for (int i = tid; i < BM * BK / 8; i += 256) {
                int r = i >> 2, k8 = (i & 3) << 3;
                cp16(&As[bf][r * BK + k8], A + (size_t)(m0 + r) * K + k0 + k8);
            }
        }
        for (int i = tid; i < BN * BK / 8; i += 256) {
            int n = i >> 2, k8 = (i & 3) << 3;
            cp16(&Bs[bf][n * BK + k8], Bt + (size_t)(n0 + n) * K + k0 + k8);
        }
    };

    AccF acc[4][4];
    #pragma unroll
    for (int mf = 0; mf < 4; ++mf)
        #pragma unroll
        for (int nf = 0; nf < 4; ++nf)
            #pragma unroll
            for (int r = 0; r < 8; ++r) acc[mf][nf].f[r] = 0.0f;

    const int nt = K / BK;
    stage(0, 0);
    for (int kt = 0; kt < nt; ++kt) {
        const int bf = kt & 1;
        cp_wait();              // async stores into buffer bf complete
        __syncthreads();        // publish buffer bf; previous compute done
        if (kt + 1 < nt) stage((kt + 1) * BK, bf ^ 1);

        BFrag a[4], b[4];
        #pragma unroll
        for (int mf = 0; mf < 4; ++mf)
            loadA(a[mf], &As[bf][(wr * 64 + mf * 16 + lq) * BK], half);
        #pragma unroll
        for (int nf = 0; nf < 4; ++nf)
            loadB(b[nf], &Bs[bf][(wc * 64 + nf * 16 + lq) * BK], half);
        #pragma unroll
        for (int mf = 0; mf < 4; ++mf)
            #pragma unroll
            for (int nf = 0; nf < 4; ++nf)
                acc[mf][nf].v = wmma_bf16(a[mf], b[nf], acc[mf][nf].v);
    }

    #pragma unroll
    for (int mf = 0; mf < 4; ++mf)
        #pragma unroll
        for (int nf = 0; nf < 4; ++nf)
            #pragma unroll
            for (int r = 0; r < 8; ++r) {
                int gm = m0 + wr * 64 + mf * 16 + r + 8 * half;
                int gn = n0 + wc * 64 + nf * 16 + lq;
                if (CMODE == 0)
                    ((unsigned short*)Cp)[(size_t)gm * N + gn] = f2u(acc[mf][nf].f[r]);
                else
                    ((float*)Cp)[(size_t)gm * N + gn] = acc[mf][nf].f[r];
            }
}

// ---------------------------------------------------------------------------
// Transpose K,V chunks: kt/vt[b][hk][c][d][s] (bf16, d-major).
// ---------------------------------------------------------------------------
__global__ __launch_bounds__(128)
void transpose_kv(const unsigned short* __restrict__ qkv,
                  unsigned short* __restrict__ kt, unsigned short* __restrict__ vt)
{
    __shared__ alignas(16) unsigned short tk[CH * DKv];
    __shared__ alignas(16) unsigned short tv[CH * DVv];
    const int c = blockIdx.x, hk = blockIdx.y, b = blockIdx.z;
    const int tid = threadIdx.x;
    const size_t rowbase = ((size_t)b * Sv + (size_t)c * CH) * QKVv;
    const int kcol = HQv * DKv + hk * DKv;
    const int vcol = HQv * DKv + HKv * DKv + hk * DVv;

    for (int i = tid; i < CH * DKv / 8; i += 128) {
        int row = i >> 3, d8 = (i & 7) << 3;
        size_t rb = rowbase + (size_t)row * QKVv;
        cp16(tk + row * DKv + d8, qkv + rb + kcol + d8);
        cp16(tv + row * DVv + d8, qkv + rb + vcol + d8);
    }
    cp_wait();
    __syncthreads();

    const size_t base = ((size_t)(b * HKv + hk) * NCv + c) * (DKv * CH);
    for (int i = tid; i < 2048; i += 128) {
        int d = i >> 5, s2 = (i & 31) << 1;
        unsigned uk = (unsigned)tk[s2 * DKv + d] | ((unsigned)tk[(s2 + 1) * DKv + d] << 16);
        unsigned uv = (unsigned)tv[s2 * DVv + d] | ((unsigned)tv[(s2 + 1) * DVv + d] << 16);
        *(unsigned*)(kt + base + d * CH + s2) = uk;
        *(unsigned*)(vt + base + d * CH + s2) = uv;
    }
}

// ---------------------------------------------------------------------------
// G[b,hk,c] = Kc^T @ Vc.  Grid (NC, HK, B), 128 threads = 4 waves.
// ---------------------------------------------------------------------------
__global__ __launch_bounds__(128)
void chunk_kv(const unsigned short* __restrict__ kt, const unsigned short* __restrict__ vt,
              unsigned short* __restrict__ g)
{
    __shared__ alignas(16) unsigned short kl[DKv * CH];
    __shared__ alignas(16) unsigned short vl[DVv * CH];
    const int c = blockIdx.x, hk = blockIdx.y, b = blockIdx.z;
    const int tid = threadIdx.x;
    const size_t base = ((size_t)(b * HKv + hk) * NCv + c) * (DKv * CH);

    for (int i = tid; i < DKv * CH / 8; i += 128) {
        cp16(&((uint4*)kl)[i], &((const uint4*)(kt + base))[i]);
        cp16(&((uint4*)vl)[i], &((const uint4*)(vt + base))[i]);
    }
    cp_wait();
    __syncthreads();

    const int w = tid >> 5, lane = tid & 31, half = lane >> 4, lq = lane & 15;
    AccF acc[4];
    #pragma unroll
    for (int nf = 0; nf < 4; ++nf)
        #pragma unroll
        for (int r = 0; r < 8; ++r) acc[nf].f[r] = 0.0f;

    #pragma unroll
    for (int ks = 0; ks < 2; ++ks) {
        BFrag a;
        loadA(a, kl + (w * 16 + lq) * CH + 32 * ks, half);
        #pragma unroll
        for (int nf = 0; nf < 4; ++nf) {
            BFrag bb;
            loadB(bb, vl + (nf * 16 + lq) * CH + 32 * ks, half);
            acc[nf].v = wmma_bf16(a, bb, acc[nf].v);
        }
    }

    const size_t gb = ((size_t)(b * HKv + hk) * NCv + c) * (DKv * DVv);
    #pragma unroll
    for (int nf = 0; nf < 4; ++nf)
        #pragma unroll
        for (int r = 0; r < 8; ++r) {
            int m = w * 16 + r + 8 * half;
            g[gb + (size_t)m * DVv + nf * 16 + lq] = f2u(acc[nf].f[r]);
        }
}

// ---------------------------------------------------------------------------
// Exclusive prefix of chunk states per q-head (+S0), stored transposed
// ([dv][dk] bf16); final f32 state to d_out tail.
// ---------------------------------------------------------------------------
__global__ __launch_bounds__(256)
void prefix_state(const unsigned short* __restrict__ g, const float* __restrict__ s0,
                  unsigned short* __restrict__ sp, float* __restrict__ sf)
{
    const int bh = blockIdx.x, b = bh >> 5, h = bh & 31, hk = h >> 2;
    const int tid = threadIdx.x;
    const size_t ibase = ((size_t)b * Hv + h) * (DKv * DVv);
    float s[16];
    #pragma unroll
    for (int j = 0; j < 16; ++j) s[j] = s0[ibase + tid + j * 256];

    for (int c = 0; c < NCv; ++c) {
        const size_t gb = ((size_t)(b * HKv + hk) * NCv + c) * (DKv * DVv);
        const size_t pb = ((size_t)(b * Hv  + h ) * NCv + c) * (DKv * DVv);
        #pragma unroll
        for (int j = 0; j < 16; ++j) {
            int e = tid + j * 256, dk = e >> 6, dv = e & 63;
            sp[pb + (size_t)dv * DKv + dk] = f2u(s[j]);
            s[j] += u2f(g[gb + e]);
        }
    }
    #pragma unroll
    for (int j = 0; j < 16; ++j) sf[ibase + tid + j * 256] = s[j];
}

// ---------------------------------------------------------------------------
// O = tril(Q K^T) V + Q S_prefix.  Grid (NC, H, B), 128 threads = 4 waves.
// ---------------------------------------------------------------------------
__global__ __launch_bounds__(128)
void attn_out(const unsigned short* __restrict__ qkv,
              const unsigned short* __restrict__ vt,
              const unsigned short* __restrict__ sp,
              unsigned short* __restrict__ o)
{
    __shared__ alignas(16) unsigned short qc[CH * DKv];
    __shared__ alignas(16) unsigned short kc[CH * DKv];
    __shared__ alignas(16) unsigned short vl[DVv * CH];
    __shared__ alignas(16) unsigned short al[CH * CH];
    __shared__ alignas(16) unsigned short sl[DVv * DKv];

    const int c = blockIdx.x, h = blockIdx.y, b = blockIdx.z, hk = h >> 2;
    const int tid = threadIdx.x;
    const size_t rowbase = ((size_t)b * Sv + (size_t)c * CH) * QKVv;
    const int qcol = h * DKv;
    const int kcol = HQv * DKv + hk * DKv;
    const size_t vbase = ((size_t)(b * HKv + hk) * NCv + c) * (DVv * CH);
    const size_t pb    = ((size_t)(b * Hv  + h ) * NCv + c) * (DKv * DVv);

    for (int i = tid; i < CH * DKv / 8; i += 128) {
        int row = i >> 3, d8 = (i & 7) << 3;
        size_t rb = rowbase + (size_t)row * QKVv;
        cp16(qc + row * DKv + d8, qkv + rb + qcol + d8);
        cp16(kc + row * DKv + d8, qkv + rb + kcol + d8);
        cp16(&((uint4*)vl)[i], &((const uint4*)(vt + vbase))[i]);
        cp16(&((uint4*)sl)[i], &((const uint4*)(sp + pb))[i]);
    }
    cp_wait();
    __syncthreads();

    const int w = tid >> 5, lane = tid & 31, half = lane >> 4, lq = lane & 15;

    // phase 1: A = Q K^T rows [16w,16w+16), causal-masked, bf16 -> al
    AccF a1[4];
    #pragma unroll
    for (int nf = 0; nf < 4; ++nf)
        #pragma unroll
        for (int r = 0; r < 8; ++r) a1[nf].f[r] = 0.0f;

    #pragma unroll
    for (int ks = 0; ks < 2; ++ks) {
        BFrag aq;
        loadA(aq, qc + (w * 16 + lq) * DKv + 32 * ks, half);
        #pragma unroll
        for (int nf = 0; nf < 4; ++nf) {
            BFrag bk;
            loadB(bk, kc + (nf * 16 + lq) * DKv + 32 * ks, half);
            a1[nf].v = wmma_bf16(aq, bk, a1[nf].v);
        }
    }
    #pragma unroll
    for (int nf = 0; nf < 4; ++nf)
        #pragma unroll
        for (int r = 0; r < 8; ++r) {
            int rowg = w * 16 + r + 8 * half;
            int colg = nf * 16 + lq;
            float vlv = (colg <= rowg) ? a1[nf].f[r] : 0.0f;
            al[rowg * CH + colg] = f2u(vlv);
        }
    __syncthreads();

    // phase 2: O cols [16w,16w+16): al@V + Q@S
    AccF a2[4];
    #pragma unroll
    for (int mf = 0; mf < 4; ++mf)
        #pragma unroll
        for (int r = 0; r < 8; ++r) a2[mf].f[r] = 0.0f;

    #pragma unroll
    for (int mf = 0; mf < 4; ++mf) {
        #pragma unroll
        for (int ks = 0; ks < 2; ++ks) {
            BFrag aa, bvf;
            loadA(aa, al + (mf * 16 + lq) * CH + 32 * ks, half);
            loadB(bvf, vl + (w * 16 + lq) * CH + 32 * ks, half);
            a2[mf].v = wmma_bf16(aa, bvf, a2[mf].v);
        }
        #pragma unroll
        for (int ks = 0; ks < 2; ++ks) {
            BFrag aq, bs;
            loadA(aq, qc + (mf * 16 + lq) * DKv + 32 * ks, half);
            loadB(bs, sl + (w * 16 + lq) * DKv + 32 * ks, half);
            a2[mf].v = wmma_bf16(aq, bs, a2[mf].v);
        }
    }

    #pragma unroll
    for (int mf = 0; mf < 4; ++mf)
        #pragma unroll
        for (int r = 0; r < 8; ++r) {
            int sg = c * CH + mf * 16 + r + 8 * half;
            size_t orow = (size_t)b * Sv + sg;
            o[orow * (Hv * DVv) + h * DVv + w * 16 + lq] = f2u(a2[mf].f[r]);
        }
}

// ---------------------------------------------------------------------------
// Host-side launch
// ---------------------------------------------------------------------------
extern "C" void kernel_launch(void* const* d_in, const int* in_sizes, int n_in,
                              void* d_out, int out_size, void* d_ws, size_t ws_size,
                              hipStream_t stream)
{
    (void)in_sizes; (void)n_in; (void)out_size; (void)ws_size;

    const float* x     = (const float*)d_in[0];
    const float* s0    = (const float*)d_in[1];
    const float* W_in  = (const float*)d_in[2];
    const float* W_out = (const float*)d_in[3];

    char* ws = (char*)d_ws;
    unsigned short* qkv   = (unsigned short*)(ws);                      //  50,331,648 B
    unsigned short* g     = (unsigned short*)(ws + (size_t)50331648);   //   8,388,608 B
    unsigned short* sp    = (unsigned short*)(ws + (size_t)58720256);   //  33,554,432 B
    unsigned short* ob    = (unsigned short*)(ws + (size_t)92274688);   //  33,554,432 B
    unsigned short* kt    = (unsigned short*)(ws + (size_t)125829120);  //   8,388,608 B
    unsigned short* vt    = (unsigned short*)(ws + (size_t)134217728);  //   8,388,608 B
    unsigned short* wtin  = (unsigned short*)(ws + (size_t)142606336);  //  12,582,912 B
    unsigned short* wtout = (unsigned short*)(ws + (size_t)155189248);  //   8,388,608 B

    float* out = (float*)d_out;
    float* sf  = out + (size_t)Bv * Sv * OUTv;

    transpose_w<<<dim3(QKVv / 32, Dv / 32), 256, 0, stream>>>(W_in, wtin, Dv, QKVv);
    transpose_w<<<dim3(OUTv / 32, (Hv * DVv) / 32), 256, 0, stream>>>(W_out, wtout, Hv * DVv, OUTv);

    gemm_wmma<0, 0><<<dim3(QKVv / 128, (Bv * Sv) / 256), 256, 0, stream>>>(
        x, wtin, qkv, Bv * Sv, QKVv, Dv);

    transpose_kv<<<dim3(NCv, HKv, Bv), 128, 0, stream>>>(qkv, kt, vt);
    chunk_kv<<<dim3(NCv, HKv, Bv), 128, 0, stream>>>(kt, vt, g);
    prefix_state<<<Bv * Hv, 256, 0, stream>>>(g, s0, sp, sf);
    attn_out<<<dim3(NCv, Hv, Bv), 128, 0, stream>>>(qkv, vt, sp, ob);

    gemm_wmma<1, 1><<<dim3(OUTv / 128, (Bv * Sv) / 256), 256, 0, stream>>>(
        ob, wtout, out, Bv * Sv, OUTv, Hv * DVv);
}
